// FusedGlobalLocalAttention_35270271434812
// MI455X (gfx1250) — matrix operand order repaired and checked by reading, not compiled, not run on hardware
//
#include <hip/hip_runtime.h>

// ---------------------------------------------------------------------------
// MI455X (gfx1250) fused global/local attention.
// All matmuls (projections, QK^T, rel logits, P*V, output proj) run on
// v_wmma_f32_16x16x32_bf16 (wave32, fp32 accumulate). K tiles are staged to
// LDS with GLOBAL_LOAD_ASYNC_TO_LDS_B128 (ASYNCcnt) when the builtin exists.
// ---------------------------------------------------------------------------

typedef __bf16 bf16;
typedef __attribute__((ext_vector_type(16))) __bf16 v16bf;
typedef __attribute__((ext_vector_type(8)))  float  v8f;
typedef __attribute__((ext_vector_type(4)))  int    i32x4;

enum : int { B_ = 2, L_ = 4096, G_ = 256, HID_ = 768, H_ = 12, D_ = 64,
             W_ = 255, N_ = 32, BLK_ = 128, HD_ = 768 };

#if defined(__has_builtin)
#if __has_builtin(__builtin_amdgcn_global_load_async_to_lds_b128) && \
    __has_builtin(__builtin_amdgcn_s_wait_asynccnt)
#define HAVE_ASYNC_LDS 1
#endif
#endif
#ifndef HAVE_ASYNC_LDS
#define HAVE_ASYNC_LDS 0
#endif

#define DEV static __device__ __forceinline__

DEV int lane_id() { return (int)(threadIdx.x & 31); }

DEV v8f v8f_zero() {
  v8f z = {0.f, 0.f, 0.f, 0.f, 0.f, 0.f, 0.f, 0.f};
  return z;
}

DEV v8f wmma_bf16(v16bf a, v16bf b, v8f c) {
  // D = A(16x32 bf16) * B(32x16 bf16) + C(16x16 f32)
  return __builtin_amdgcn_wmma_f32_16x16x32_bf16(false, a, false, b,
                                                 (short)0, c, false, false);
}

// 16-byte global -> LDS copy; async (ASYNCcnt) on gfx1250 toolchains that
// expose the builtin, otherwise synchronous through VGPRs.
DEV void cp16_g2lds(void* lds_dst, const void* gsrc) {
#if HAVE_ASYNC_LDS
  __builtin_amdgcn_global_load_async_to_lds_b128(
      (__attribute__((address_space(1))) i32x4*)(void*)gsrc,
      (__attribute__((address_space(3))) i32x4*)lds_dst, 0, 0);
#else
  *reinterpret_cast<uint4*>(lds_dst) = *reinterpret_cast<const uint4*>(gsrc);
#endif
}

DEV void cp_wait() {
#if HAVE_ASYNC_LDS
  __builtin_amdgcn_s_wait_asynccnt(0);
#endif
}

// A fragment (16M x 32K bf16) from row-major [M][K] slab; p already at (row0,k0).
// Layout: lanes 0-15 rows M, K 0..15; lanes 16-31 K 16..31 (matches load_bt).
DEV v16bf load_a(const bf16* p, int ld) {
  const int l = lane_id();
  const bf16* r = p + (l & 15) * ld + ((l < 16) ? 0 : 16);
  v16bf a;
#pragma unroll
  for (int j = 0; j < 16; ++j) a[j] = r[j];
  return a;
}

// B fragment (32K x 16N bf16) read from B^T stored row-major [N][K]; p at (n0,k0).
// Layout used here (contiguous K, same as load_a): lanes 0-15 hold K 0..15, lanes 16-31 hold K 16..31, col = lane&15.
DEV v16bf load_bt(const bf16* p, int ld) {
  const int l = lane_id();
  const bf16* r = p + (l & 15) * ld + ((l < 16) ? 0 : 16);
  v16bf b;
#pragma unroll
  for (int j = 0; j < 16; ++j) b[j] = r[j];
  return b;
}

DEV v16bf load_bt_f32(const float* p, int ld) {
  const int l = lane_id();
  const float* r = p + (l & 15) * ld + ((l < 16) ? 0 : 16);
  v16bf b;
#pragma unroll
  for (int j = 0; j < 16; ++j) b[j] = (bf16)r[j];
  return b;
}

// ---------------------------------------------------------------------------
// Generic GEMM: Y[M,N] = X[M,K] @ W[K,N] + bias, bf16 WMMA compute.
// Workgroup: 256 threads = 8 waves (4x2), tile 128M x 64N, K step 32.
// ---------------------------------------------------------------------------
template <typename InT, typename OutT>
__global__ void __launch_bounds__(256)
gemm_bias(const InT* __restrict__ X, const float* __restrict__ Wm,
          const float* __restrict__ bias, OutT* __restrict__ Y,
          int M, int K, int N) {
  __shared__ __align__(16) bf16 As[128][40];
  __shared__ __align__(16) bf16 Bt[64][40];   // transposed: [n][k]

  const int tid  = threadIdx.x;
  const int wave = tid >> 5;
  const int wm   = wave & 3;       // 0..3 -> 32-row slice
  const int wn   = wave >> 2;      // 0..1 -> 32-col slice
  const int tiles_n = N >> 6;
  const int row0 = (blockIdx.x / tiles_n) * 128;
  const int col0 = (blockIdx.x % tiles_n) * 64;

  v8f a00 = v8f_zero(), a01 = v8f_zero(), a10 = v8f_zero(), a11 = v8f_zero();

  for (int k0 = 0; k0 < K; k0 += 32) {
    __syncthreads();
    { // A tile 128x32 (convert to bf16)
      const int r = tid >> 3;
      const int c = (tid & 7) * 4;
#pragma unroll
      for (int pp = 0; pp < 4; ++pp) {
        const int row = r + pp * 32;
        const InT* src = X + (size_t)(row0 + row) * K + k0 + c;
        bf16 t0, t1, t2, t3;
        if constexpr (sizeof(InT) == 4) {
          const float4 v = *reinterpret_cast<const float4*>(src);
          t0 = (bf16)v.x; t1 = (bf16)v.y; t2 = (bf16)v.z; t3 = (bf16)v.w;
        } else {
          t0 = (bf16)src[0]; t1 = (bf16)src[1]; t2 = (bf16)src[2]; t3 = (bf16)src[3];
        }
        As[row][c + 0] = t0; As[row][c + 1] = t1;
        As[row][c + 2] = t2; As[row][c + 3] = t3;
      }
    }
    { // B tile 32x64, stored transposed into Bt[n][k]
#pragma unroll
      for (int pp = 0; pp < 2; ++pp) {
        const int lin = pp * 256 + tid;
        const int kr  = lin >> 4;
        const int c   = (lin & 15) * 4;
        const float4 v =
            *reinterpret_cast<const float4*>(Wm + (size_t)(k0 + kr) * N + col0 + c);
        Bt[c + 0][kr] = (bf16)v.x; Bt[c + 1][kr] = (bf16)v.y;
        Bt[c + 2][kr] = (bf16)v.z; Bt[c + 3][kr] = (bf16)v.w;
      }
    }
    __syncthreads();
    const v16bf fa0 = load_a(&As[wm * 32][0], 40);
    const v16bf fa1 = load_a(&As[wm * 32 + 16][0], 40);
    const v16bf fb0 = load_bt(&Bt[wn * 32][0], 40);
    const v16bf fb1 = load_bt(&Bt[wn * 32 + 16][0], 40);
    a00 = wmma_bf16(fa0, fb0, a00);
    a01 = wmma_bf16(fa0, fb1, a01);
    a10 = wmma_bf16(fa1, fb0, a10);
    a11 = wmma_bf16(fa1, fb1, a11);
  }

  const int l = lane_id();
  const int n = l & 15, hi = l >> 4;
#pragma unroll
  for (int mi = 0; mi < 2; ++mi)
#pragma unroll
    for (int ni = 0; ni < 2; ++ni) {
      const v8f acc = (mi == 0) ? (ni == 0 ? a00 : a01) : (ni == 0 ? a10 : a11);
      const int col = col0 + wn * 32 + ni * 16 + n;
      const float bv = bias[col];
#pragma unroll
      for (int r = 0; r < 8; ++r) {
        const int row = row0 + wm * 32 + mi * 16 + r + hi * 8;
        Y[(size_t)row * N + col] = (OutT)(acc[r] + bv);
      }
    }
}

// ---------------------------------------------------------------------------
// Local blocked attention: one WG per (b, block n, head h); 8 waves x 16 rows.
// Keys: 3*BLK local window (skew-expanded) + G global side keys = 640,
// streamed through LDS in 20 chunks of 32 with online softmax.
// ---------------------------------------------------------------------------
__global__ void __launch_bounds__(256)
local_attn(const bf16* __restrict__ lq, const bf16* __restrict__ lk,
           const bf16* __restrict__ lv, const bf16* __restrict__ gk,
           const bf16* __restrict__ gv,
           const int* __restrict__ l2l_mask, const int* __restrict__ l2g_mask,
           const int* __restrict__ l2l_ids,  const int* __restrict__ l2g_ids,
           const float* __restrict__ rel_emb, const float* __restrict__ rel_bias,
           bf16* __restrict__ ctx) {
  __shared__ __align__(16) float relall[128][33];
  __shared__ __align__(16) bf16  Kt[32][72];      // [key][d]  (B^T for QK^T)
  __shared__ __align__(16) bf16  Vt[64][40];      // [d][key]  (B^T for P*V)
  __shared__ __align__(16) bf16  Pst[8][16][36];  // per-wave P staging

  const int tid = threadIdx.x;
  const int w   = tid >> 5;
  const int l   = tid & 31;
  const int n   = l & 15, hi = l >> 4;

  const int h  = (int)(blockIdx.x % H_);
  const int nb = (int)((blockIdx.x / H_) % N_);
  const int b  = (int)(blockIdx.x / (H_ * N_));

  const int q0 = w * 16;  // wave's query-row base within the 128-token block
  const bf16* qptr = lq + ((size_t)b * L_ + (size_t)nb * BLK_ + q0) * HD_ + h * D_;
  const v16bf aq0 = load_a(qptr, HD_);       // d 0..31
  const v16bf aq1 = load_a(qptr + 32, HD_);  // d 32..63

  { // rel_all[q][v] = Q . rel_emb[v,h,:] + rel_bias[v,h]
    const float* rb = rel_emb + h * D_;      // element (v,d): rb + v*768 + d
    v8f c0 = v8f_zero(), c1 = v8f_zero();
    c0 = wmma_bf16(aq0, load_bt_f32(rb, H_ * D_), c0);
    c0 = wmma_bf16(aq1, load_bt_f32(rb + 32, H_ * D_), c0);
    c1 = wmma_bf16(aq0, load_bt_f32(rb + 16 * H_ * D_, H_ * D_), c1);
    c1 = wmma_bf16(aq1, load_bt_f32(rb + 16 * H_ * D_ + 32, H_ * D_), c1);
#pragma unroll
    for (int r = 0; r < 8; ++r) {
      const int ql = q0 + r + hi * 8;
      relall[ql][n]      = c0[r] + rel_bias[n * H_ + h];
      relall[ql][16 + n] = c1[r] + rel_bias[(16 + n) * H_ + h];
    }
  }

  float m_r[8], l_r[8];
  v8f acc[4];
#pragma unroll
  for (int r = 0; r < 8; ++r) { m_r[r] = -1e30f; l_r[r] = 0.f; }
#pragma unroll
  for (int t = 0; t < 4; ++t) acc[t] = v8f_zero();

  for (int ch = 0; ch < 20; ++ch) {           // 640 keys / 32
    __syncthreads();                          // prior P*V reads done
    { // cooperative K/V chunk load (32 keys x 64 d), zero-padded out of range
      const int key = tid >> 3;
      const int dg  = (tid & 7) * 8;
      const int j   = ch * 32 + key;
      const bf16* ksrc = nullptr;
      const bf16* vsrc = nullptr;
      if (j >= 3 * BLK_) {
        const int g = j - 3 * BLK_;
        const size_t o = ((size_t)b * G_ + g) * HD_ + h * D_ + dg;
        ksrc = gk + o; vsrc = gv + o;
      } else {
        const int tok = (nb - 1) * BLK_ + j;
        if (tok >= 0 && tok < L_) {
          const size_t o = ((size_t)b * L_ + tok) * HD_ + h * D_ + dg;
          ksrc = lk + o; vsrc = lv + o;
        }
      }
      if (ksrc) {
        cp16_g2lds(&Kt[key][dg], ksrc);       // async global->LDS (ASYNCcnt)
        union { uint4 u; bf16 e[8]; } vu;
        vu.u = *reinterpret_cast<const uint4*>(vsrc);
#pragma unroll
        for (int jj = 0; jj < 8; ++jj) Vt[dg + jj][key] = vu.e[jj];
      } else {
        const uint4 z = make_uint4(0, 0, 0, 0);
        *reinterpret_cast<uint4*>(&Kt[key][dg]) = z;
#pragma unroll
        for (int jj = 0; jj < 8; ++jj) Vt[dg + jj][key] = (bf16)0.f;
      }
      // streaming prefetch two chunks ahead (single-use data)
      const int jn = j + 64;
      if (jn < 640) {
        const bf16* pf = nullptr;
        if (jn >= 3 * BLK_) {
          pf = gk + ((size_t)b * G_ + (jn - 3 * BLK_)) * HD_ + h * D_ + dg;
        } else {
          const int tokn = (nb - 1) * BLK_ + jn;
          if (tokn >= 0 && tokn < L_)
            pf = lk + ((size_t)b * L_ + tokn) * HD_ + h * D_ + dg;
        }
        if (pf) __builtin_prefetch(pf, 0, 1);
      }
    }
    cp_wait();                                // s_wait_asynccnt 0
    __syncthreads();

    float sv[2][8];
#pragma unroll
    for (int st = 0; st < 2; ++st) {          // two 16-key subtiles
      v8f s = v8f_zero();
      s = wmma_bf16(aq0, load_bt(&Kt[st * 16][0], 72), s);
      s = wmma_bf16(aq1, load_bt(&Kt[st * 16][32], 72), s);
      const int j = ch * 32 + st * 16 + n;    // column (key) index, 0..639
#pragma unroll
      for (int r = 0; r < 8; ++r) {
        const int ql = q0 + r + hi * 8;       // query pos in block
        const int qg = nb * BLK_ + ql;        // absolute token
        int id = 0; float mk = 0.f;
        if (j >= 3 * BLK_) {
          const size_t o = ((size_t)b * L_ + qg) * G_ + (j - 3 * BLK_);
          id = l2g_ids[o]; mk = (float)l2g_mask[o];
        } else {
          const int rr  = j - ql - 1;         // skew index into 2R+1 window
          const int tok = (nb - 1) * BLK_ + j;
          if (rr >= 0 && rr < W_ && tok >= 0 && tok < L_) {
            const size_t o = ((size_t)b * L_ + qg) * W_ + rr;
            id = l2l_ids[o]; mk = (float)l2l_mask[o];
          }
        }
        sv[st][r] = (s[r] + relall[ql][id]) * 0.125f - 10000.f * (1.f - mk);
      }
    }

    float fr[8];
#pragma unroll
    for (int r = 0; r < 8; ++r) {             // online softmax per row
      float t = fmaxf(sv[0][r], sv[1][r]);
#pragma unroll
      for (int x = 1; x < 16; x <<= 1) t = fmaxf(t, __shfl_xor(t, x, 32));
      const float nm = fmaxf(m_r[r], t);
      fr[r] = __expf(m_r[r] - nm);
      m_r[r] = nm;
      const float p0 = __expf(sv[0][r] - nm);
      const float p1 = __expf(sv[1][r] - nm);
      sv[0][r] = p0; sv[1][r] = p1;
      float rs = p0 + p1;
#pragma unroll
      for (int x = 1; x < 16; x <<= 1) rs += __shfl_xor(rs, x, 32);
      l_r[r] = l_r[r] * fr[r] + rs;
    }
#pragma unroll
    for (int r = 0; r < 8; ++r) {             // stage P (bf16) + rescale O
      const int mrow = r + hi * 8;
      Pst[w][mrow][n]      = (bf16)sv[0][r];
      Pst[w][mrow][16 + n] = (bf16)sv[1][r];
#pragma unroll
      for (int t = 0; t < 4; ++t) acc[t][r] = acc[t][r] * fr[r];
    }
    __syncthreads();
    const v16bf ap = load_a(&Pst[w][0][0], 36);
#pragma unroll
    for (int t = 0; t < 4; ++t)
      acc[t] = wmma_bf16(ap, load_bt(&Vt[t * 16][0], 40), acc[t]);
  }

#pragma unroll
  for (int t = 0; t < 4; ++t)
#pragma unroll
    for (int r = 0; r < 8; ++r) {
      const int ql = q0 + r + hi * 8;
      const float val = acc[t][r] / l_r[r];
      ctx[((size_t)b * L_ + (size_t)nb * BLK_ + ql) * HD_ + h * D_ + t * 16 + n] =
          (bf16)val;
    }
}

// ---------------------------------------------------------------------------
// Global attention: one WG per (b, h, 64-query tile); 4 waves x 16 rows.
// Keys: [gk(256); lk(4096)] = 4352, streamed in 136 chunks of 32.
// ---------------------------------------------------------------------------
__global__ void __launch_bounds__(128)
glob_attn(const bf16* __restrict__ gq, const bf16* __restrict__ gk,
          const bf16* __restrict__ gv, const bf16* __restrict__ lk,
          const bf16* __restrict__ lv,
          const int* __restrict__ g2g_mask, const int* __restrict__ g2l_mask,
          const int* __restrict__ g2g_ids,  const int* __restrict__ g2l_ids,
          const float* __restrict__ rel_emb, const float* __restrict__ rel_bias,
          bf16* __restrict__ ctx) {
  __shared__ __align__(16) float relall[64][33];
  __shared__ __align__(16) bf16  Kt[32][72];
  __shared__ __align__(16) bf16  Vt[64][40];
  __shared__ __align__(16) bf16  Pst[4][16][36];

  const int tid = threadIdx.x;
  const int w   = tid >> 5;
  const int l   = tid & 31;
  const int n   = l & 15, hi = l >> 4;

  const int qt = (int)(blockIdx.x & 3);
  const int h  = (int)((blockIdx.x >> 2) % H_);
  const int b  = (int)(blockIdx.x / (4 * H_));

  const int q0 = qt * 64 + w * 16;            // wave's global-query row base
  const bf16* qptr = gq + ((size_t)b * G_ + q0) * HD_ + h * D_;
  const v16bf aq0 = load_a(qptr, HD_);
  const v16bf aq1 = load_a(qptr + 32, HD_);

  {
    const float* rb = rel_emb + h * D_;
    v8f c0 = v8f_zero(), c1 = v8f_zero();
    c0 = wmma_bf16(aq0, load_bt_f32(rb, H_ * D_), c0);
    c0 = wmma_bf16(aq1, load_bt_f32(rb + 32, H_ * D_), c0);
    c1 = wmma_bf16(aq0, load_bt_f32(rb + 16 * H_ * D_, H_ * D_), c1);
    c1 = wmma_bf16(aq1, load_bt_f32(rb + 16 * H_ * D_ + 32, H_ * D_), c1);
#pragma unroll
    for (int r = 0; r < 8; ++r) {
      const int qr = w * 16 + r + hi * 8;     // row within this WG's 64 rows
      relall[qr][n]      = c0[r] + rel_bias[n * H_ + h];
      relall[qr][16 + n] = c1[r] + rel_bias[(16 + n) * H_ + h];
    }
  }

  float m_r[8], l_r[8];
  v8f acc[4];
#pragma unroll
  for (int r = 0; r < 8; ++r) { m_r[r] = -1e30f; l_r[r] = 0.f; }
#pragma unroll
  for (int t = 0; t < 4; ++t) acc[t] = v8f_zero();

  for (int ch = 0; ch < (G_ + L_) / 32; ++ch) {
    __syncthreads();
    { // K/V chunk load: 128 threads, each 1 key x 16 d
      const int key = tid >> 2;
      const int dg  = (tid & 3) * 16;
      const int j   = ch * 32 + key;
      const bf16 *ks, *vs;
      if (j < G_) {
        const size_t o = ((size_t)b * G_ + j) * HD_ + h * D_ + dg;
        ks = gk + o; vs = gv + o;
      } else {
        const size_t o = ((size_t)b * L_ + (j - G_)) * HD_ + h * D_ + dg;
        ks = lk + o; vs = lv + o;
      }
      cp16_g2lds(&Kt[key][dg],     ks);       // async global->LDS
      cp16_g2lds(&Kt[key][dg + 8], ks + 8);
      union { uint4 u; bf16 e[8]; } v0, v1;
      v0.u = reinterpret_cast<const uint4*>(vs)[0];
      v1.u = reinterpret_cast<const uint4*>(vs)[1];
#pragma unroll
      for (int jj = 0; jj < 8; ++jj) {
        Vt[dg + jj][key]     = v0.e[jj];
        Vt[dg + 8 + jj][key] = v1.e[jj];
      }
    }
    cp_wait();
    __syncthreads();

    float sv[2][8];
#pragma unroll
    for (int st = 0; st < 2; ++st) {
      v8f s = v8f_zero();
      s = wmma_bf16(aq0, load_bt(&Kt[st * 16][0], 72), s);
      s = wmma_bf16(aq1, load_bt(&Kt[st * 16][32], 72), s);
      const int j = ch * 32 + st * 16 + n;
#pragma unroll
      for (int r = 0; r < 8; ++r) {
        const int q = q0 + r + hi * 8;        // absolute global query 0..255
        int id; float mk;
        if (j < G_) {
          const size_t o = ((size_t)b * G_ + q) * G_ + j;
          id = g2g_ids[o]; mk = (float)g2g_mask[o];
        } else {
          const size_t o = ((size_t)b * G_ + q) * L_ + (j - G_);
          id = g2l_ids[o]; mk = (float)g2l_mask[o];
        }
        sv[st][r] = (s[r] + relall[q - qt * 64][id]) * 0.125f - 10000.f * (1.f - mk);
      }
    }

    float fr[8];
#pragma unroll
    for (int r = 0; r < 8; ++r) {
      float t = fmaxf(sv[0][r], sv[1][r]);
#pragma unroll
      for (int x = 1; x < 16; x <<= 1) t = fmaxf(t, __shfl_xor(t, x, 32));
      const float nm = fmaxf(m_r[r], t);
      fr[r] = __expf(m_r[r] - nm);
      m_r[r] = nm;
      const float p0 = __expf(sv[0][r] - nm);
      const float p1 = __expf(sv[1][r] - nm);
      sv[0][r] = p0; sv[1][r] = p1;
      float rs = p0 + p1;
#pragma unroll
      for (int x = 1; x < 16; x <<= 1) rs += __shfl_xor(rs, x, 32);
      l_r[r] = l_r[r] * fr[r] + rs;
    }
#pragma unroll
    for (int r = 0; r < 8; ++r) {
      const int mrow = r + hi * 8;
      Pst[w][mrow][n]      = (bf16)sv[0][r];
      Pst[w][mrow][16 + n] = (bf16)sv[1][r];
#pragma unroll
      for (int t = 0; t < 4; ++t) acc[t][r] = acc[t][r] * fr[r];
    }
    __syncthreads();
    const v16bf ap = load_a(&Pst[w][0][0], 36);
#pragma unroll
    for (int t = 0; t < 4; ++t)
      acc[t] = wmma_bf16(ap, load_bt(&Vt[t * 16][0], 40), acc[t]);
  }

#pragma unroll
  for (int t = 0; t < 4; ++t)
#pragma unroll
    for (int r = 0; r < 8; ++r) {
      const int q = q0 + r + hi * 8;
      const float val = acc[t][r] / l_r[r];
      ctx[((size_t)b * G_ + q) * HD_ + h * D_ + t * 16 + n] = (bf16)val;
    }
}

// ---------------------------------------------------------------------------
extern "C" void kernel_launch(void* const* d_in, const int* in_sizes, int n_in,
                              void* d_out, int out_size, void* d_ws, size_t ws_size,
                              hipStream_t stream) {
  (void)in_sizes; (void)n_in; (void)out_size; (void)ws_size;

  const float* long_input = (const float*)d_in[0];
  const float* glob_input = (const float*)d_in[1];
  const int* l2l_mask = (const int*)d_in[2];
  const int* g2g_mask = (const int*)d_in[3];
  const int* l2g_mask = (const int*)d_in[4];
  const int* g2l_mask = (const int*)d_in[5];
  const int* l2l_ids  = (const int*)d_in[6];
  const int* g2g_ids  = (const int*)d_in[7];
  const int* l2g_ids  = (const int*)d_in[8];
  const int* g2l_ids  = (const int*)d_in[9];
  const float* wq_long = (const float*)d_in[10]; const float* bq_long = (const float*)d_in[11];
  const float* wk_long = (const float*)d_in[12]; const float* bk_long = (const float*)d_in[13];
  const float* wv_long = (const float*)d_in[14]; const float* bv_long = (const float*)d_in[15];
  const float* wq_glob = (const float*)d_in[16]; const float* bq_glob = (const float*)d_in[17];
  const float* wk_glob = (const float*)d_in[18]; const float* bk_glob = (const float*)d_in[19];
  const float* wv_glob = (const float*)d_in[20]; const float* bv_glob = (const float*)d_in[21];
  const float* rel_emb_long  = (const float*)d_in[22];
  const float* rel_bias_long = (const float*)d_in[23];
  const float* rel_emb_glob  = (const float*)d_in[24];
  const float* rel_bias_glob = (const float*)d_in[25];
  const float* wo_long = (const float*)d_in[26]; const float* bo_long = (const float*)d_in[27];
  const float* wo_glob = (const float*)d_in[28]; const float* bo_glob = (const float*)d_in[29];

  char* ws = (char*)d_ws;
  auto take = [&](size_t bytes) -> char* {
    char* p = ws;
    ws += (bytes + 255) & ~(size_t)255;
    return p;
  };
  const size_t lbytes = (size_t)B_ * L_ * HD_ * sizeof(bf16);
  const size_t gbytes = (size_t)B_ * G_ * HD_ * sizeof(bf16);
  bf16* lqb  = (bf16*)take(lbytes);
  bf16* lkb  = (bf16*)take(lbytes);
  bf16* lvb  = (bf16*)take(lbytes);
  bf16* gqb  = (bf16*)take(gbytes);
  bf16* gkb  = (bf16*)take(gbytes);
  bf16* gvb  = (bf16*)take(gbytes);
  bf16* ctxl = (bf16*)take(lbytes);
  bf16* ctxg = (bf16*)take(gbytes);

  const int gridL = (B_ * L_ / 128) * (HD_ / 64);  // 768
  const int gridG = (B_ * G_ / 128) * (HD_ / 64);  // 48

  // input projections (fp32 -> bf16 q/k/v)
  gemm_bias<float, bf16><<<gridL, 256, 0, stream>>>(long_input, wq_long, bq_long, lqb, B_ * L_, HID_, HD_);
  gemm_bias<float, bf16><<<gridL, 256, 0, stream>>>(long_input, wk_long, bk_long, lkb, B_ * L_, HID_, HD_);
  gemm_bias<float, bf16><<<gridL, 256, 0, stream>>>(long_input, wv_long, bv_long, lvb, B_ * L_, HID_, HD_);
  gemm_bias<float, bf16><<<gridG, 256, 0, stream>>>(glob_input, wq_glob, bq_glob, gqb, B_ * G_, HID_, HD_);
  gemm_bias<float, bf16><<<gridG, 256, 0, stream>>>(glob_input, wk_glob, bk_glob, gkb, B_ * G_, HID_, HD_);
  gemm_bias<float, bf16><<<gridG, 256, 0, stream>>>(glob_input, wv_glob, bv_glob, gvb, B_ * G_, HID_, HD_);

  // attention
  local_attn<<<B_ * N_ * H_, 256, 0, stream>>>(lqb, lkb, lvb, gkb, gvb,
                                               l2l_mask, l2g_mask, l2l_ids, l2g_ids,
                                               rel_emb_long, rel_bias_long, ctxl);
  glob_attn<<<B_ * H_ * 4, 128, 0, stream>>>(gqb, gkb, gvb, lkb, lvb,
                                             g2g_mask, g2l_mask, g2g_ids, g2l_ids,
                                             rel_emb_glob, rel_bias_glob, ctxg);

  // output projections (bf16 ctx -> fp32 out)
  float* out_long = (float*)d_out;
  float* out_glob = out_long + (size_t)B_ * L_ * HID_;
  gemm_bias<bf16, float><<<gridL, 256, 0, stream>>>(ctxl, wo_long, bo_long, out_long, B_ * L_, HD_, HID_);
  gemm_bias<bf16, float><<<gridG, 256, 0, stream>>>(ctxg, wo_glob, bo_glob, out_glob, B_ * G_, HD_, HID_);
}
